// PolarSpectralLoss_38869454029284
// MI455X (gfx1250) — matrix operand order repaired and checked by reading, not compiled, not run on hardware
//
#include <hip/hip_runtime.h>
#include <hip/hip_bf16.h>
#include <math.h>

// ---------------------------------------------------------------------------
// PolarSpectralLoss on MI455X (gfx1250, wave32, WMMA 16x16x32 f16->f32)
//
// 2D FFT as dense DFT-matrix GEMMs on the WMMA pipes with CDNA5 data movers:
//   gemm1: global_load_async_to_lds_b128 (ASYNCcnt) double-buffered staging
//   gemm2: TENSOR_LOAD_TO_LDS (TDM, TENSORcnt) tile DMA with HW LDS padding
// Fragments feed WMMA from LDS via ds_load_b128.
//
//   Zr  = L*Fr                 Zi' = L*S          (row transform)
//   Sr  = Fr*Zr - S*Zi'        |Si| = Fr*Zi' + S*Zr   (col transform)
// Second GEMM fuses fftshift + log1p(|Sp|) + polar sector pooling.
// ---------------------------------------------------------------------------

typedef __attribute__((ext_vector_type(16))) _Float16 v16h;
typedef __attribute__((ext_vector_type(8)))  _Float16 v8h;
typedef __attribute__((ext_vector_type(8)))  float    v8f;

#define LD 512
#define PI_F 3.14159265358979323846f

// LDS tile row stride in halves: 32 data + 8 pad = 40 halves = 80 B.
// 16-B aligned rows for ds_load_b128; 20-bank stride -> conflict-free
// 16-row fragment reads (r*20 mod 64 distinct for r=0..15).
#define T_STRIDE 40

// ---- CDNA5 async copy: global -> LDS, 16 bytes per lane --------------------
#define ASYNC_CP16(ldsp, gp)                                                   \
    asm volatile("global_load_async_to_lds_b128 %0, %1, off"                   \
                 :: "v"((unsigned int)(size_t)(ldsp)), "v"((const void*)(gp))  \
                 : "memory")

#define WAIT_ASYNC() asm volatile("s_wait_asynccnt 0x0" ::: "memory")

// ---- CDNA5 Tensor Data Mover (D# per CDNA5 ISA) ------------
#if defined(__HIP_DEVICE_COMPILE__) && __has_builtin(__builtin_amdgcn_tensor_load_to_lds)
#define HAVE_TDM 1
#endif

#ifdef HAVE_TDM
typedef unsigned int u32x4 __attribute__((ext_vector_type(4)));
typedef int          i32x8 __attribute__((ext_vector_type(8)));
typedef int          i32x4 __attribute__((ext_vector_type(4)));

// 2-D tile load from a 512x512 f16 tensor (row stride 512 elements) into LDS,
// hardware-padding each 32-element row (64 B) by 4 DWORDs -> T_STRIDE rows.
// This toolchain exposes the 6-arg builtin: (g0, g1, g2, g3, g4, cpol).
__device__ __forceinline__ void tdm_load_tile(const _Float16* gsrc,
                                              unsigned lds_off,
                                              unsigned tile_w, unsigned tile_h) {
    unsigned long long ga = (unsigned long long)(size_t)gsrc;
    u32x4 g0;
    g0[0] = 1u;                                            // count=1 (user D#)
    g0[1] = lds_off;                                       // lds_addr
    g0[2] = (unsigned)ga;                                  // global_addr[31:0]
    g0[3] = (unsigned)((ga >> 32) & 0x1FFFFFFu) | (2u << 30); // addr[56:32], type=2
    i32x8 g1;
    g1[0] = (int)((1u << 16)        // data_size = 2 B
                | (1u << 20)        // pad_enable
                | (3u << 22)        // pad_interval: 16 DWORDs (64 B) of data
                | (3u << 25));      // pad_amount:   4 DWORDs (16 B)
    g1[1] = (int)(512u << 16);      // tensor_dim0[15:0] @ [63:48]
    g1[2] = (int)(512u << 16);      // tensor_dim0 hi=0, tensor_dim1[15:0]
    g1[3] = (int)(tile_w << 16);    // tensor_dim1 hi=0, tile_dim0
    g1[4] = (int)tile_h;            // tile_dim1; tile_dim2 = 0
    g1[5] = 512;                    // tensor_dim0_stride[31:0]
    g1[6] = (int)(512u << 16);      // stride0 hi=0, tensor_dim1_stride[15:0]
    g1[7] = 0;
    i32x4 gz4 = {0, 0, 0, 0};                 // groups 2/3 unused (2-D tile)
    i32x8 gz8 = {0, 0, 0, 0, 0, 0, 0, 0};     // extra group (6-arg form)
    __builtin_amdgcn_tensor_load_to_lds(g0, g1, gz4, gz4, gz8, 0);
}
#define WAIT_G2() __builtin_amdgcn_s_wait_tensorcnt(0)
#else
#define WAIT_G2() WAIT_ASYNC()
#endif

// ---- WMMA fragment helpers (layouts per CDNA5 ISA 7.12.2, wave32) ----------

__device__ __forceinline__ v16h lds_a_frag(const _Float16* tile, int mrow0, int lane) {
    int half = lane >> 4;
    int row  = mrow0 + (lane & 15);
    const _Float16* p = tile + row * T_STRIDE + (half << 3);
    v8h lo = *(const v8h*)(p);
    v8h hi = *(const v8h*)(p + 16);
    v16h f;
#pragma unroll
    for (int i = 0; i < 8; ++i) { f[i] = lo[i]; f[i + 8] = hi[i]; }
    return f;
}

// B tile is 32 rows x 32 cols; nt selects the 16-column n-subtile.
__device__ __forceinline__ v16h lds_b_frag(const _Float16* tile, int lane, int nt) {
    int half = lane >> 4;
    const _Float16* p = tile + (half << 3) * T_STRIDE + (nt << 4) + (lane & 15);
    v16h f;
#pragma unroll
    for (int i = 0; i < 8; ++i) { f[i] = p[i * T_STRIDE]; f[i + 8] = p[(i + 16) * T_STRIDE]; }
    return f;
}

__device__ __forceinline__ v8f wmma_f16(v16h a, v16h b, v8f c) {
    return __builtin_amdgcn_wmma_f32_16x16x32_f16(false, a, false, b, (short)0, c,
                                                  false, false);
}

// ---- polar sector binning --------------------------------------------------

__device__ __forceinline__ int sector_bin(int i, int j) {
    float yy = fmaf((float)i, 2.0f / 511.0f, -1.0f);
    float xx = fmaf((float)j, 2.0f / 511.0f, -1.0f);
    float rad = sqrtf(xx * xx + yy * yy) * 0.70710678118f;
    float ang = atan2f(yy, xx);
    ang = fmodf(ang, PI_F);
    if (ang < 0.0f) ang += PI_F;
    int ab = (int)(ang * (8.0f / PI_F));
    ab = ab < 0 ? 0 : (ab > 7 ? 7 : ab);
    if (rad < 0.08f || rad > 0.9f) return -1;
    int rb = (int)((rad - 0.08f) * (3.0f / 0.82f));
    rb = rb < 0 ? 0 : (rb > 2 ? 2 : rb);
    return rb * 8 + ab;
}

// ---- kernel 1: luma + Hann window -> f16 -----------------------------------

__global__ void __launch_bounds__(256) prep_luma(const float* __restrict__ pred,
                                                 const float* __restrict__ tgt,
                                                 _Float16* __restrict__ Lh) {
    int idx = blockIdx.x * 256 + threadIdx.x;
    int x = idx & 511;
    int y = (idx >> 9) & 511;
    int b = idx >> 18;
    const float* src = (b < 16) ? (pred + (size_t)b * 3 * 262144)
                                : (tgt  + (size_t)(b - 16) * 3 * 262144);
    size_t o = (size_t)y * 512 + x;
    float r = src[o];
    float g = src[o + 262144];
    float bl = src[o + 524288];
    float l = 0.2989f * r + 0.587f * g + 0.114f * bl;
    float wy = 0.5f * (1.0f - cosf((2.0f * PI_F / 511.0f) * (float)y));
    float wx = 0.5f * (1.0f - cosf((2.0f * PI_F / 511.0f) * (float)x));
    Lh[idx] = (_Float16)(l * wy * wx);
}

// ---- kernel 2: DFT matrices (exact twiddle index) --------------------------

__global__ void __launch_bounds__(256) dft_mats(_Float16* __restrict__ Fr,
                                                _Float16* __restrict__ Sm) {
    int idx = blockIdx.x * 256 + threadIdx.x;
    int i = idx >> 9;
    int j = idx & 511;
    int t = (i * j) & 511;
    float a = (2.0f * PI_F / 512.0f) * (float)t;
    Fr[idx] = (_Float16)cosf(a);
    Sm[idx] = (_Float16)sinf(a);
}

// ---- kernel 3: zero accumulators + mask pixel counts -----------------------

__global__ void __launch_bounds__(256) zero_and_mask(float* __restrict__ accg,
                                                     float* __restrict__ cnt) {
    __shared__ float c[24];
    int t = threadIdx.x;
    if (t < 24) c[t] = 0.0f;
    for (int i = t; i < 32 * 24; i += 256) accg[i] = 0.0f;
    __syncthreads();
    for (int idx = t; idx < 262144; idx += 256) {
        int bin = sector_bin(idx >> 9, idx & 511);
        if (bin >= 0) atomicAdd(&c[bin], 1.0f);
    }
    __syncthreads();
    if (t < 24) cnt[t] = c[t];
}

// ---- kernel 4: row transform  Zr = L*Fr, Zi' = L*S -------------------------
// Block: 8 waves -> 128x32 output slab (each wave: 1 m-tile x 2 n-tiles).
// A(128x32) + B(32x32 x2) double buffered in LDS via per-lane async copies.

__global__ void __launch_bounds__(256) gemm1(const _Float16* __restrict__ Lh,
                                             const _Float16* __restrict__ Fr,
                                             const _Float16* __restrict__ Sm,
                                             _Float16* __restrict__ Zr,
                                             _Float16* __restrict__ Zi) {
    __shared__ _Float16 sA [2][128 * T_STRIDE];
    __shared__ _Float16 sBr[2][32 * T_STRIDE];
    __shared__ _Float16 sBs[2][32 * T_STRIDE];

    int t    = threadIdx.x;
    int lane = t & 31;
    int w    = t >> 5;
    int bx   = blockIdx.x;                 // 2048 blocks
    int batch = bx >> 6;                   // 64 blocks per image
    int rem   = bx & 63;
    int mbase = (rem >> 4) << 7;           // 4 m-blocks of 128 rows
    int n0    = (rem & 15) << 5;           // 16 n-blocks of 32 cols
    const _Float16* A = Lh + (size_t)batch * 262144;

    auto stage = [&](int buf, int k0) {
        // A tile: 128 rows x 32 halves = 512 x 16-B chunks; 2 per thread.
#pragma unroll
        for (int i = 0; i < 2; ++i) {
            int c = t + (i << 8);
            int row = c >> 2, cc = c & 3;
            ASYNC_CP16(&sA[buf][row * T_STRIDE + cc * 8],
                       A + (size_t)(mbase + row) * LD + k0 + cc * 8);
        }
        // B tiles: 32 rows x 32 halves = 128 chunks each; waves 0-3 -> Fr,
        // waves 4-7 -> Sm (uniform per wave).
        {
            int c = t & 127;
            int row = c >> 2, cc = c & 3;
            const _Float16* src = (t < 128) ? Fr : Sm;
            _Float16* dst = (t < 128) ? &sBr[buf][0] : &sBs[buf][0];
            ASYNC_CP16(dst + row * T_STRIDE + cc * 8,
                       src + (size_t)(k0 + row) * LD + n0 + cc * 8);
        }
    };

    stage(0, 0);
    WAIT_ASYNC();
    __syncthreads();

    v8f cr0 = {}, cr1 = {}, ci0 = {}, ci1 = {};
    for (int step = 0; step < 16; ++step) {
        int cur = step & 1;
        if (step < 15) stage(cur ^ 1, (step + 1) << 5);
        v16h a   = lds_a_frag(sA[cur], w << 4, lane);
        v16h br0 = lds_b_frag(sBr[cur], lane, 0);
        v16h br1 = lds_b_frag(sBr[cur], lane, 1);
        v16h bs0 = lds_b_frag(sBs[cur], lane, 0);
        v16h bs1 = lds_b_frag(sBs[cur], lane, 1);
        cr0 = wmma_f16(a, br0, cr0);
        cr1 = wmma_f16(a, br1, cr1);
        ci0 = wmma_f16(a, bs0, ci0);
        ci1 = wmma_f16(a, bs1, ci1);
        if (step < 15) { WAIT_ASYNC(); __syncthreads(); }
    }

    int col   = n0 + (lane & 15);
    int rbase = mbase + (w << 4) + ((lane >> 4) << 3);
    _Float16* pr = Zr + (size_t)batch * 262144;
    _Float16* pi = Zi + (size_t)batch * 262144;
#pragma unroll
    for (int v = 0; v < 8; ++v) {
        size_t ro = (size_t)(rbase + v) * LD;
        pr[ro + col]      = (_Float16)cr0[v];
        pr[ro + col + 16] = (_Float16)cr1[v];
        pi[ro + col]      = (_Float16)ci0[v];
        pi[ro + col + 16] = (_Float16)ci1[v];
    }
}

// ---- kernel 5: col transform + fused log-magnitude polar pooling -----------
// Staging via Tensor Data Mover: 4 descriptors/step (Fr, S, Zr, Zi), one per
// wave 0..3; synchronized with s_wait_tensorcnt + workgroup barrier.

__global__ void __launch_bounds__(256) gemm2(const _Float16* __restrict__ Fr,
                                             const _Float16* __restrict__ Sm,
                                             const _Float16* __restrict__ Zr,
                                             const _Float16* __restrict__ Zi,
                                             float* __restrict__ accg) {
    __shared__ _Float16 sAr[2][128 * T_STRIDE];
    __shared__ _Float16 sAs[2][128 * T_STRIDE];
    __shared__ _Float16 sBr[2][32 * T_STRIDE];
    __shared__ _Float16 sBi[2][32 * T_STRIDE];
    __shared__ float bins[24];

    int t = threadIdx.x;
    if (t < 24) bins[t] = 0.0f;

    int lane = t & 31;
    int w    = t >> 5;
    int bx   = blockIdx.x;
    int batch = bx >> 6;
    int rem   = bx & 63;
    int mbase = (rem >> 4) << 7;
    int n0    = (rem & 15) << 5;
    const _Float16* Br = Zr + (size_t)batch * 262144;
    const _Float16* Bi = Zi + (size_t)batch * 262144;

#ifdef HAVE_TDM
    auto stage = [&](int buf, int k0) {
        if (w == 0)
            tdm_load_tile(Fr + (size_t)mbase * LD + k0,
                          (unsigned)(size_t)&sAr[buf][0], 32, 128);
        else if (w == 1)
            tdm_load_tile(Sm + (size_t)mbase * LD + k0,
                          (unsigned)(size_t)&sAs[buf][0], 32, 128);
        else if (w == 2)
            tdm_load_tile(Br + (size_t)k0 * LD + n0,
                          (unsigned)(size_t)&sBr[buf][0], 32, 32);
        else if (w == 3)
            tdm_load_tile(Bi + (size_t)k0 * LD + n0,
                          (unsigned)(size_t)&sBi[buf][0], 32, 32);
    };
#else
    auto stage = [&](int buf, int k0) {
#pragma unroll
        for (int i = 0; i < 2; ++i) {
            int c = t + (i << 8);
            int row = c >> 2, cc = c & 3;
            size_t go = (size_t)(mbase + row) * LD + k0 + cc * 8;
            int    lo = row * T_STRIDE + cc * 8;
            ASYNC_CP16(&sAr[buf][lo], Fr + go);
            ASYNC_CP16(&sAs[buf][lo], Sm + go);
        }
        {
            int c = t & 127;
            int row = c >> 2, cc = c & 3;
            const _Float16* src = (t < 128) ? Br : Bi;
            _Float16* dst = (t < 128) ? &sBr[buf][0] : &sBi[buf][0];
            ASYNC_CP16(dst + row * T_STRIDE + cc * 8,
                       src + (size_t)(k0 + row) * LD + n0 + cc * 8);
        }
    };
#endif

    stage(0, 0);
    WAIT_G2();
    __syncthreads();

    v8f p0 = {}, p1 = {};     // Fr*Zr
    v8f q0 = {}, q1 = {};     // S*Zi'
    v8f s0 = {}, s1 = {};     // Fr*Zi' + S*Zr
    for (int step = 0; step < 16; ++step) {
        int cur = step & 1;
        if (step < 15) stage(cur ^ 1, (step + 1) << 5);
        v16h afr  = lds_a_frag(sAr[cur], w << 4, lane);
        v16h afs  = lds_a_frag(sAs[cur], w << 4, lane);
        v16h bzr0 = lds_b_frag(sBr[cur], lane, 0);
        v16h bzr1 = lds_b_frag(sBr[cur], lane, 1);
        v16h bzi0 = lds_b_frag(sBi[cur], lane, 0);
        v16h bzi1 = lds_b_frag(sBi[cur], lane, 1);
        p0 = wmma_f16(afr, bzr0, p0);
        p1 = wmma_f16(afr, bzr1, p1);
        q0 = wmma_f16(afs, bzi0, q0);
        q1 = wmma_f16(afs, bzi1, q1);
        s0 = wmma_f16(afr, bzi0, s0);
        s0 = wmma_f16(afs, bzr0, s0);
        s1 = wmma_f16(afr, bzi1, s1);
        s1 = wmma_f16(afs, bzr1, s1);
        if (step < 15) { WAIT_G2(); __syncthreads(); }
    }

    int col   = n0 + (lane & 15);
    int rbase = mbase + (w << 4) + ((lane >> 4) << 3);
#pragma unroll
    for (int v = 0; v < 8; ++v) {
        int i = (rbase + v + 256) & 511;           // fftshift row
        {
            float sr = p0[v] - q0[v];
            float mag = log1pf(sqrtf(sr * sr + s0[v] * s0[v]));
            int bin = sector_bin(i, (col + 256) & 511);
            if (bin >= 0) atomicAdd(&bins[bin], mag);
        }
        {
            float sr = p1[v] - q1[v];
            float mag = log1pf(sqrtf(sr * sr + s1[v] * s1[v]));
            int bin = sector_bin(i, (col + 16 + 256) & 511);
            if (bin >= 0) atomicAdd(&bins[bin], mag);
        }
    }
    __syncthreads();
    if (t < 24) atomicAdd(&accg[batch * 24 + t], bins[t]);
}

// ---- kernel 6: energies, normalization, Charbonnier, weighted mean ---------

__global__ void __launch_bounds__(768) final_reduce(const float* __restrict__ accg,
                                                    const float* __restrict__ cnt,
                                                    const float* __restrict__ w,
                                                    float* __restrict__ out) {
    __shared__ float e[768];
    __shared__ float rowsum[96];
    __shared__ float red[768];
    int t = threadIdx.x;
    float v = accg[t] / fmaxf(cnt[t % 24], 1e-6f);
    e[t] = v;
    __syncthreads();
    if (t < 96) {
        float s = 0.0f;
#pragma unroll
        for (int i = 0; i < 8; ++i) s += e[t * 8 + i];
        rowsum[t] = fmaxf(s, 1e-6f);
    }
    __syncthreads();
    float en = e[t] / rowsum[t >> 3];
    __syncthreads();
    e[t] = en;
    __syncthreads();
    float d = 0.0f;
    if (t < 384) {
        float diff = e[t] - e[t + 384];
        int r = (t >> 3) % 3;
        d = sqrtf(diff * diff + 1e-6f) * w[r];
    }
    red[t] = d;
    __syncthreads();
    for (int s = 384; s > 0; s >>= 1) {
        if (t < s) red[t] += red[t + s];
        __syncthreads();
    }
    if (t == 0) out[0] = red[0] / 384.0f;
}

// ---------------------------------------------------------------------------

extern "C" void kernel_launch(void* const* d_in, const int* in_sizes, int n_in,
                              void* d_out, int out_size, void* d_ws, size_t ws_size,
                              hipStream_t stream) {
    const float* pred = (const float*)d_in[0];
    const float* tgt  = (const float*)d_in[1];
    const float* w    = (const float*)d_in[2];

    char* ws = (char*)d_ws;
    _Float16* Lh = (_Float16*)(ws);                  // 16,777,216 B
    _Float16* Fr = (_Float16*)(ws + 16777216);       //    524,288 B
    _Float16* Sm = (_Float16*)(ws + 17301504);       //    524,288 B
    _Float16* Zr = (_Float16*)(ws + 17825792);       // 16,777,216 B
    _Float16* Zi = (_Float16*)(ws + 34603008);       // 16,777,216 B
    float*    ac = (float*)   (ws + 51380224);       //      3,072 B
    float*    ct = (float*)   (ws + 51383296);       //         96 B

    prep_luma    <<<32768, 256, 0, stream>>>(pred, tgt, Lh);
    dft_mats     <<<1024,  256, 0, stream>>>(Fr, Sm);
    zero_and_mask<<<1,     256, 0, stream>>>(ac, ct);
    gemm1        <<<2048,  256, 0, stream>>>(Lh, Fr, Sm, Zr, Zi);
    gemm2        <<<2048,  256, 0, stream>>>(Fr, Sm, Zr, Zi, ac);
    final_reduce <<<1,     768, 0, stream>>>(ac, ct, w, (float*)d_out);
}
